// LlamaAttention_6588479832091
// MI455X (gfx1250) — matrix operand order repaired and checked by reading, not compiled, not run on hardware
//
#include <hip/hip_runtime.h>
#include <hip/hip_bf16.h>
#include <math.h>
#include <stdint.h>

// ---------------------------------------------------------------------------
// LlamaAttention for MI455X (gfx1250): bf16 WMMA pipeline, fp32 accumulate,
// TDM (tensor_load_to_lds) staging for GEMM/attention tiles.
//   B=4, S=4096, HIDDEN=2048, H=16, HD=128
// ---------------------------------------------------------------------------

#define B_    4
#define S_    4096
#define HID_  2048
#define H_    16
#define HD_   128
#define M_    (B_ * S_)     // 16384 rows

typedef __bf16 bf16_t;
typedef __bf16 bf16x8  __attribute__((ext_vector_type(8)));
typedef __bf16 bf16x16 __attribute__((ext_vector_type(16)));
typedef float  f32x8   __attribute__((ext_vector_type(8)));

union BF16Frag { bf16x16 v; bf16x8 h[2]; };

__device__ inline f32x8 f32x8_zero() {
  f32x8 z = {0.f, 0.f, 0.f, 0.f, 0.f, 0.f, 0.f, 0.f};
  return z;
}

// ---------------------------------------------------------------------------
// Tensor Data Mover: 2D tile load global -> LDS (ISA 08_async_tensor D#).
// Descriptor built per ISA 8.3/8.4; pad feature reproduces our bank-skew pads.
// ---------------------------------------------------------------------------
#ifdef __gfx1250__
#  if __has_builtin(__builtin_amdgcn_tensor_load_to_lds)
#    define TDM_AVAILABLE 1
#  endif
#endif
#ifndef TDM_AVAILABLE
#  define TDM_AVAILABLE 0
#endif

#if TDM_AVAILABLE
typedef unsigned int u32x4 __attribute__((ext_vector_type(4)));
typedef int          i32x4 __attribute__((ext_vector_type(4)));
typedef int          i32x8 __attribute__((ext_vector_type(8)));

// width_elems x rows bf16 tile, source row stride = stride_elems.
// pad_iv: LDS pad interval code (interval = 2^(code+1) dwords)
// pad_am: LDS pad amount  code (amount  = code+1     dwords)
__device__ inline void tdm_load_2d_bf16(const void* gptr, unsigned lds_off,
                                        unsigned width_elems, unsigned rows,
                                        unsigned stride_elems,
                                        unsigned pad_iv, unsigned pad_am,
                                        unsigned pad_en) {
  unsigned long long ga = (unsigned long long)(uintptr_t)gptr;
  u32x4 g0;
  g0[0] = 1u;                                        // count=1 (valid descriptor)
  g0[1] = lds_off;                                   // lds_addr (bytes)
  g0[2] = (unsigned)ga;                              // global_addr[31:0]
  g0[3] = (unsigned)((ga >> 32) & 0x1FFFFFFull)      // global_addr[56:32]
        | (2u << 30);                                // type=2 ("image")
  i32x8 g1;
  g1[0] = (int)((1u << 16)                           // data_size=1 -> 2 bytes
              | (pad_en << 20) | (pad_iv << 22) | (pad_am << 25));
  g1[1] = (int)(width_elems << 16);                  // tensor_dim0[15:0]
  g1[2] = (int)((width_elems >> 16) | (rows << 16)); // tensor_dim0[31:16]|tensor_dim1[15:0]
  g1[3] = (int)((rows >> 16) | (width_elems << 16)); // tensor_dim1[31:16]|tile_dim0
  g1[4] = (int)(rows & 0xFFFFu);                     // tile_dim1 (tile_dim2 = 0)
  g1[5] = (int)stride_elems;                         // tensor_dim0_stride[31:0]
  g1[6] = 0;                                         // stride[47:32]|dim1_stride[15:0]
  g1[7] = 0;
  i32x4 z4 = {0, 0, 0, 0};
#if __clang_major__ >= 23
  i32x8 z8 = {0, 0, 0, 0, 0, 0, 0, 0};
  __builtin_amdgcn_tensor_load_to_lds(g0, g1, z4, z4, z8, 0);
#else
  __builtin_amdgcn_tensor_load_to_lds(g0, g1, z4, z4, 0);
#endif
}
__device__ inline void tdm_wait0() { __builtin_amdgcn_s_wait_tensorcnt(0); }
__device__ inline unsigned lds_off_u32(const void* p) {
  return (unsigned)(uintptr_t)p;   // LDS aperture: low 32 bits = byte offset
}
#endif  // TDM_AVAILABLE

// ---------------------------------------------------------------------------
// Cast fp32 -> bf16 (vectorized x4)
// ---------------------------------------------------------------------------
__global__ __launch_bounds__(256) void cast_f32_to_bf16(
    const float* __restrict__ src, bf16_t* __restrict__ dst, long n) {
  long i = ((long)blockIdx.x * blockDim.x + threadIdx.x) * 4;
  if (i + 3 < n) {
    float4 f = *(const float4*)(src + i);
    dst[i + 0] = (bf16_t)f.x;
    dst[i + 1] = (bf16_t)f.y;
    dst[i + 2] = (bf16_t)f.z;
    dst[i + 3] = (bf16_t)f.w;
  }
}

// ---------------------------------------------------------------------------
// RoPE: out[..., j]    = x[2j]*cos - x[2j+1]*sin     (j in [0,64))
//       out[..., j+64] = x[2j]*sin + x[2j+1]*cos
// ---------------------------------------------------------------------------
__global__ __launch_bounds__(256) void rope_bf16(
    const bf16_t* __restrict__ qraw, const bf16_t* __restrict__ kraw,
    const float* __restrict__ inv_freq,
    bf16_t* __restrict__ qout, bf16_t* __restrict__ kout) {
  long i  = (long)blockIdx.x * blockDim.x + threadIdx.x;  // [0, M*H*64)
  int  j  = (int)(i & 63);
  int  hh = (int)((i >> 6) & (H_ - 1));
  long m  = i >> 10;
  int  s  = (int)(m & (S_ - 1));
  float sn, cs;
  sincosf((float)s * inv_freq[j], &sn, &cs);
  long base = m * HID_ + (long)hh * HD_;
  {
    float x1 = (float)qraw[base + 2 * j];
    float x2 = (float)qraw[base + 2 * j + 1];
    qout[base + j]      = (bf16_t)(x1 * cs - x2 * sn);
    qout[base + 64 + j] = (bf16_t)(x1 * sn + x2 * cs);
  }
  {
    float x1 = (float)kraw[base + 2 * j];
    float x2 = (float)kraw[base + 2 * j + 1];
    kout[base + j]      = (bf16_t)(x1 * cs - x2 * sn);
    kout[base + 64 + j] = (bf16_t)(x1 * sn + x2 * cs);
  }
}

// ---------------------------------------------------------------------------
// GEMM: C[M,N] = A[M,K] @ W[N,K]^T, bf16 in, WMMA f32 acc.
// Block tile 128x128, BK=32, 256 threads = 8 waves, wave tile 32x64.
// Tiles staged into double-buffered padded LDS by TDM (wave 0 issues the D#,
// waits TENSORcnt, workgroup barrier publishes). Fallback: VGPR staging.
// ---------------------------------------------------------------------------
template <bool OUT_F32>
__global__ __launch_bounds__(256) void gemm_xwT_wmma(
    const bf16_t* __restrict__ A, const bf16_t* __restrict__ W,
    void* __restrict__ Cout, int M, int N, int K) {
  constexpr int BK = 32, BKP = 40;  // 16-dword rows + 4-dword pad (TDM pad 3/3)
  __shared__ bf16_t As[2][128][BKP];
  __shared__ bf16_t Ws[2][128][BKP];

  const int tid  = threadIdx.x;
  const int wave = tid >> 5, lane = tid & 31;
  const int lrow = lane & 15, half = lane >> 4;
  const int wm = wave >> 1, wn = wave & 1;
  const long rowBase = (long)blockIdx.y * 128;
  const long colBase = (long)blockIdx.x * 128;

  f32x8 acc[2][4];
  for (int i = 0; i < 2; ++i)
    for (int j = 0; j < 4; ++j) acc[i][j] = f32x8_zero();

  const int KT = K / BK;

#if TDM_AVAILABLE
  const unsigned ldsA[2] = {lds_off_u32(&As[0][0][0]), lds_off_u32(&As[1][0][0])};
  const unsigned ldsW[2] = {lds_off_u32(&Ws[0][0][0]), lds_off_u32(&Ws[1][0][0])};
  auto stage = [&](int kt, int buf) {
    const long k0 = (long)kt * BK;
    // 128 rows x 32 bf16; pad: every 16 dwords insert 4 dwords (code 3 / 3).
    tdm_load_2d_bf16(A + rowBase * (long)K + k0, ldsA[buf], BK, 128, K, 3, 3, 1);
    tdm_load_2d_bf16(W + colBase * (long)K + k0, ldsW[buf], BK, 128, K, 3, 3, 1);
  };
  if (wave == 0) { stage(0, 0); tdm_wait0(); }
#else
  const int lr = tid >> 2;
  const int lk = (tid & 3) * 8;
  auto stage = [&](int kt, int buf) {
    const long k0 = (long)kt * BK + lk;
    const bf16_t* ap = A + (rowBase + lr) * (long)K + k0;
    const bf16_t* wp = W + (colBase + lr) * (long)K + k0;
    *(bf16x8*)&As[buf][lr][lk]      = *(const bf16x8*)ap;
    *(bf16x8*)&As[buf][lr + 64][lk] = *(const bf16x8*)(ap + 64L * K);
    *(bf16x8*)&Ws[buf][lr][lk]      = *(const bf16x8*)wp;
    *(bf16x8*)&Ws[buf][lr + 64][lk] = *(const bf16x8*)(wp + 64L * K);
    __builtin_prefetch(ap + 2 * BK, 0, 1);
    __builtin_prefetch(wp + 2 * BK, 0, 1);
  };
  stage(0, 0);
#endif
  __syncthreads();

  for (int kt = 0; kt < KT; ++kt) {
    const int cur = kt & 1;
#if TDM_AVAILABLE
    if (wave == 0 && kt + 1 < KT) stage(kt + 1, cur ^ 1);  // DMA overlaps compute
#else
    if (kt + 1 < KT) stage(kt + 1, cur ^ 1);
#endif

    BF16Frag af[2];
    for (int mt = 0; mt < 2; ++mt) {
      const int r = wm * 32 + mt * 16 + lrow;
      af[mt].h[0] = *(const bf16x8*)&As[cur][r][8 * half];
      af[mt].h[1] = *(const bf16x8*)&As[cur][r][16 + 8 * half];
    }
    for (int nt = 0; nt < 4; ++nt) {
      const int c = wn * 64 + nt * 16 + lrow;
      BF16Frag bfr;
      bfr.h[0] = *(const bf16x8*)&Ws[cur][c][8 * half];
      bfr.h[1] = *(const bf16x8*)&Ws[cur][c][16 + 8 * half];
      for (int mt = 0; mt < 2; ++mt)
        acc[mt][nt] = __builtin_amdgcn_wmma_f32_16x16x32_bf16(
            false, af[mt].v, false, bfr.v, (short)0, acc[mt][nt], false, false);
    }
#if TDM_AVAILABLE
    if (wave == 0) tdm_wait0();  // next tile landed before we publish
#endif
    __syncthreads();
  }

  for (int mt = 0; mt < 2; ++mt)
    for (int nt = 0; nt < 4; ++nt)
      for (int r = 0; r < 8; ++r) {
        long row = rowBase + wm * 32 + mt * 16 + r + 8 * half;
        long col = colBase + wn * 64 + nt * 16 + lrow;
        float v = acc[mt][nt][r];
        if (OUT_F32) ((float*)Cout)[row * N + col] = v;
        else         ((bf16_t*)Cout)[row * N + col] = (bf16_t)v;
      }
}

// ---------------------------------------------------------------------------
// Flash attention (non-causal): one block per (b, h, 128-row query tile).
// K tile staged by TDM (row-major + pad); V tile transposed manually (TDM has
// no transpose). Q fragments pinned in registers; online softmax in registers
// with __shfl_xor over the 16-lane half that owns each row.
// ---------------------------------------------------------------------------
__global__ __launch_bounds__(256) void flash_attn_wmma(
    const bf16_t* __restrict__ Q, const bf16_t* __restrict__ Kc,
    const bf16_t* __restrict__ V, bf16_t* __restrict__ O) {
  constexpr int QB = 128, KB = 64;
  const float scale = 0.08838834764831845f;  // 1/sqrt(128)

  __shared__ bf16_t Ks[KB][HD_ + 8];   // 64-dword rows + 4-dword pad (TDM 5/3)
  __shared__ bf16_t Vt[HD_][KB + 8];
  __shared__ bf16_t Ps[QB][KB + 8];
  __shared__ float  mrun[QB], lrun[QB];

  const int tid  = threadIdx.x;
  const int w    = tid >> 5, lane = tid & 31;
  const int lrow = lane & 15, half = lane >> 4;
  const int b = blockIdx.y >> 4, h = blockIdx.y & 15;
  const long rowBase = (long)b * S_ + (long)blockIdx.x * QB;
  const int  headOff = h * HD_;

  if (tid < QB) { mrun[tid] = -INFINITY; lrun[tid] = 0.f; }

  BF16Frag qf[4];
  {
    const bf16_t* qp = Q + (rowBase + w * 16 + lrow) * (long)HID_ + headOff;
    for (int kk = 0; kk < 4; ++kk) {
      qf[kk].h[0] = *(const bf16x8*)(qp + kk * 32 + 8 * half);
      qf[kk].h[1] = *(const bf16x8*)(qp + kk * 32 + 16 + 8 * half);
    }
  }

  f32x8 o[8];
  for (int i = 0; i < 8; ++i) o[i] = f32x8_zero();

  const int kr = tid >> 2;         // key row this thread handles for V
  const int d0 = (tid & 3) * 32;

#if TDM_AVAILABLE
  const unsigned ldsKs = lds_off_u32(&Ks[0][0]);
#endif

  for (int j = 0; j < S_ / KB; ++j) {
    __syncthreads();  // previous iteration's readers are done with Ks/Vt
#if TDM_AVAILABLE
    if (w == 0)       // 64 keys x 128 dims; pad every 64 dwords by 4 dwords
      tdm_load_2d_bf16(Kc + ((long)(b * S_ + j * KB)) * HID_ + headOff,
                       ldsKs, HD_, KB, HID_, 5, 3, 1);
#else
    {
      const bf16_t* kp = Kc + ((long)(b * S_ + j * KB + kr)) * HID_ + headOff + d0;
      for (int c = 0; c < 4; ++c)
        *(bf16x8*)&Ks[kr][d0 + c * 8] = *(const bf16x8*)(kp + c * 8);
    }
#endif
    {
      const bf16_t* vp = V + ((long)(b * S_ + j * KB + kr)) * HID_ + headOff + d0;
      for (int c = 0; c < 4; ++c) {
        bf16x8 vv = *(const bf16x8*)(vp + c * 8);
        for (int e = 0; e < 8; ++e) Vt[d0 + c * 8 + e][kr] = vv[e];
      }
    }
#if TDM_AVAILABLE
    if (w == 0) tdm_wait0();
#endif
    __syncthreads();

    // ---- scores: S = Q x K^T  (16 x 64 per wave) ----
    f32x8 sc[4];
    for (int nt = 0; nt < 4; ++nt) {
      f32x8 a = f32x8_zero();
      for (int kk = 0; kk < 4; ++kk) {
        BF16Frag kbf;
        kbf.h[0] = *(const bf16x8*)&Ks[nt * 16 + lrow][kk * 32 + 8 * half];
        kbf.h[1] = *(const bf16x8*)&Ks[nt * 16 + lrow][kk * 32 + 16 + 8 * half];
        a = __builtin_amdgcn_wmma_f32_16x16x32_bf16(
            false, qf[kk].v, false, kbf.v, (short)0, a, false, false);
      }
      for (int r = 0; r < 8; ++r) a[r] *= scale;
      sc[nt] = a;
    }

    // ---- online softmax ----
    float mblk[8];
    for (int r = 0; r < 8; ++r)
      mblk[r] = fmaxf(fmaxf(sc[0][r], sc[1][r]), fmaxf(sc[2][r], sc[3][r]));
    for (int off = 1; off < 16; off <<= 1)
      for (int r = 0; r < 8; ++r)
        mblk[r] = fmaxf(mblk[r], __shfl_xor(mblk[r], off, 32));

    float mnew[8], efac[8];
    for (int r = 0; r < 8; ++r) {
      const int row = w * 16 + r + 8 * half;
      const float mo = mrun[row];
      mnew[r] = fmaxf(mo, mblk[r]);
      efac[r] = __expf(mo - mnew[r]);
    }

    float lblk[8] = {0, 0, 0, 0, 0, 0, 0, 0};
    for (int nt = 0; nt < 4; ++nt)
      for (int r = 0; r < 8; ++r) {
        float p = __expf(sc[nt][r] - mnew[r]);
        lblk[r] += p;
        Ps[w * 16 + r + 8 * half][nt * 16 + lrow] = (bf16_t)p;
      }
    for (int off = 1; off < 16; off <<= 1)
      for (int r = 0; r < 8; ++r)
        lblk[r] += __shfl_xor(lblk[r], off, 32);
    if (lrow == 0)
      for (int r = 0; r < 8; ++r) {
        const int row = w * 16 + r + 8 * half;
        mrun[row] = mnew[r];
        lrun[row] = lrun[row] * efac[r] + lblk[r];
      }

    // ---- rescale O, then O += P x V ----
    for (int dt = 0; dt < 8; ++dt)
      for (int r = 0; r < 8; ++r) o[dt][r] *= efac[r];

    for (int kk = 0; kk < 2; ++kk) {
      BF16Frag pa;  // same-wave LDS RAW: DS pipe is in-order per wave
      pa.h[0] = *(const bf16x8*)&Ps[w * 16 + lrow][kk * 32 + 8 * half];
      pa.h[1] = *(const bf16x8*)&Ps[w * 16 + lrow][kk * 32 + 16 + 8 * half];
      for (int dt = 0; dt < 8; ++dt) {
        BF16Frag vbf;
        vbf.h[0] = *(const bf16x8*)&Vt[dt * 16 + lrow][kk * 32 + 8 * half];
        vbf.h[1] = *(const bf16x8*)&Vt[dt * 16 + lrow][kk * 32 + 16 + 8 * half];
        o[dt] = __builtin_amdgcn_wmma_f32_16x16x32_bf16(
            false, pa.v, false, vbf.v, (short)0, o[dt], false, false);
      }
    }
  }

  float inv[8];
  for (int r = 0; r < 8; ++r) inv[r] = 1.f / lrun[w * 16 + r + 8 * half];
  for (int dt = 0; dt < 8; ++dt)
    for (int r = 0; r < 8; ++r) {
      const long row = rowBase + w * 16 + r + 8 * half;
      O[row * HID_ + headOff + dt * 16 + lrow] = (bf16_t)(o[dt][r] * inv[r]);
    }
}

// ---------------------------------------------------------------------------
// Host-side launch: cast -> Q/K/V GEMMs -> RoPE -> flash attention -> out GEMM
// ---------------------------------------------------------------------------
extern "C" void kernel_launch(void* const* d_in, const int* in_sizes, int n_in,
                              void* d_out, int out_size, void* d_ws, size_t ws_size,
                              hipStream_t stream) {
  (void)in_sizes; (void)n_in; (void)out_size; (void)ws_size;
  const float* x    = (const float*)d_in[0];
  const float* wq   = (const float*)d_in[1];
  const float* wk   = (const float*)d_in[2];
  const float* wv   = (const float*)d_in[3];
  const float* wo   = (const float*)d_in[4];
  const float* invf = (const float*)d_in[5];

  const long mh = (long)M_ * HID_;
  const long ww = (long)HID_ * HID_;

  char*   ws   = (char*)d_ws;
  bf16_t* xb   = (bf16_t*)ws;          // x bf16 (reused as roped q)
  bf16_t* wqb  = xb + mh;
  bf16_t* wkb  = wqb + ww;
  bf16_t* wvb  = wkb + ww;
  bf16_t* wob  = wvb + ww;
  bf16_t* qraw = wob + ww;             // q pre-RoPE (reused as attn out)
  bf16_t* kraw = qraw + mh;
  bf16_t* kb   = kraw + mh;
  bf16_t* vb   = kb + mh;
  bf16_t* qb   = xb;
  bf16_t* ab   = qraw;

  dim3 blk(256);
  cast_f32_to_bf16<<<(unsigned)(mh / 1024), blk, 0, stream>>>(x, xb, mh);
  cast_f32_to_bf16<<<(unsigned)(ww / 1024), blk, 0, stream>>>(wq, wqb, ww);
  cast_f32_to_bf16<<<(unsigned)(ww / 1024), blk, 0, stream>>>(wk, wkb, ww);
  cast_f32_to_bf16<<<(unsigned)(ww / 1024), blk, 0, stream>>>(wv, wvb, ww);
  cast_f32_to_bf16<<<(unsigned)(ww / 1024), blk, 0, stream>>>(wo, wob, ww);

  dim3 ggrid(HID_ / 128, M_ / 128);
  gemm_xwT_wmma<false><<<ggrid, blk, 0, stream>>>(xb, wqb, qraw, M_, HID_, HID_);
  gemm_xwT_wmma<false><<<ggrid, blk, 0, stream>>>(xb, wkb, kraw, M_, HID_, HID_);
  gemm_xwT_wmma<false><<<ggrid, blk, 0, stream>>>(xb, wvb, vb,   M_, HID_, HID_);

  const long rth = (long)M_ * H_ * 64;
  rope_bf16<<<(unsigned)(rth / 256), blk, 0, stream>>>(qraw, kraw, invf, qb, kb);

  dim3 agrid(S_ / 128, B_ * H_);
  flash_attn_wmma<<<agrid, blk, 0, stream>>>(qb, kb, vb, ab);

  gemm_xwT_wmma<true><<<ggrid, blk, 0, stream>>>(ab, wob, d_out, M_, HID_, HID_);
}
